// GIPRB_42236708389342
// MI455X (gfx1250) — matrix operand order repaired and checked by reading, not compiled, not run on hardware
//
#include <hip/hip_runtime.h>
#include <hip/hip_bf16.h>
#include <math.h>

// ---------------------------------------------------------------------------
// GIPRB fused block for MI455X (gfx1250).
// One workgroup (256 threads = 8 wave32) per batch element; all intermediates
// in 7 LDS slots of 224x32 f32 (padded M) + an 18KB f16 B-fragment staging
// buffer (219,136 B dynamic LDS; WGP allows 320KB). Every linear / conv is
// implicit-GEMM on v_wmma_f32_16x16x32_f16:
//   - A fragments: vectorized float4 LDS reads (dense) or branchless
//     clamp+select gathers (im2col) -> unconditional, batchable ds loads.
//   - B fragments: weights converted once per GEMM into LDS f16 laid out in
//     the exact per-lane WMMA fragment order -> one 32B LDS read per WMMA.
// Input / output cross HBM via GLOBAL_{LOAD,STORE}_ASYNC_*_LDS_B128
// (ASYNCcnt-tracked, coalesced 16B transfers). Attention (head dim 4) is VALU.
// ---------------------------------------------------------------------------

typedef __attribute__((ext_vector_type(16))) _Float16 v16h;
typedef __attribute__((ext_vector_type(8)))  float    v8f;

#define CCH 32
#define TT  9
#define LL  24
#define PP  216                 // t*l positions
#define NTHREADS 256
#define NW  8                   // waves per block
#define MROWS 224               // PP padded to 14 M-tiles
#define SLOT (MROWS*CCH)        // 7168 floats per slot
#define BST_HALFS (18*512)      // max Ntiles*Ksteps = 18 fragments
#define SMEM_BYTES (7*SLOT*4 + BST_HALFS*2)   // 219,136 B dynamic LDS

struct Params {
  const float *dsw,*dsb,*dsg,*dsbeta;
  const float *lp1w,*lp1b,*lp2w,*lp2b,*mw,*mb;
  const float *aqw,*aqb,*akw,*akb,*avw,*avb,*amw,*amb;
  const float *aprompt,*apcw,*apcb,*apcg,*apcbeta,*aplw,*aplb;
  const float *iqw,*iqb,*ikw,*ikb,*ivw,*ivb,*imw,*imb;
  const float *iprompt,*ipcw,*ipcb,*ipcg,*ipcbeta,*iplw,*iplb;
  const float *c1w0,*c1b0,*c1g0,*c1beta0;
  const float *c1w1,*c1b1,*c1g1,*c1beta1;
  const float *c2w0,*c2b0,*c2g0,*c2beta0;
  const float *c2w1,*c2b1,*c2g1,*c2beta1;
  const float *fqw,*fqb,*fkw,*fkb,*fvw,*fvb,*fmw,*fmb;
  const int   *flag;
  const float *x;
  float       *out;
};

// --- async global <-> LDS (CDNA5, ASYNCcnt) -------------------------------
__device__ inline unsigned lds_off(const void* p) {   // low 32b = LDS offset
  return (unsigned)(unsigned long long)p;
}
__device__ inline void async_g2l_b128(unsigned ldsaddr, const void* gaddr) {
  asm volatile("global_load_async_to_lds_b128 %0, %1, off"
               :: "v"(ldsaddr), "v"(gaddr) : "memory");
}
__device__ inline void async_l2g_b128(void* gaddr, unsigned ldsaddr) {
  asm volatile("global_store_async_from_lds_b128 %0, %1, off"
               :: "v"(gaddr), "v"(ldsaddr) : "memory");
}
__device__ inline void wait_async0() {
  asm volatile("s_wait_asynccnt 0x0" ::: "memory");
}

// --- A fragment, dense row-major [m][32] f32, vectorized ------------------
// Layout (ISA 7.12.2): lane half h: elems 0..7 -> K=h*8+e, 8..15 -> K=16+h*8+e.
__device__ inline v16h load_a_rm(const float* S, int m0, int lane) {
  const float* r = S + (m0 + (lane & 15)) * CCH + ((lane >> 4) << 3);
  const float4 x0 = *(const float4*)(r);
  const float4 x1 = *(const float4*)(r + 4);
  const float4 x2 = *(const float4*)(r + 16);
  const float4 x3 = *(const float4*)(r + 20);
  v16h a;
  a[0]=(_Float16)x0.x;  a[1]=(_Float16)x0.y;  a[2]=(_Float16)x0.z;  a[3]=(_Float16)x0.w;
  a[4]=(_Float16)x1.x;  a[5]=(_Float16)x1.y;  a[6]=(_Float16)x1.z;  a[7]=(_Float16)x1.w;
  a[8]=(_Float16)x2.x;  a[9]=(_Float16)x2.y;  a[10]=(_Float16)x2.z; a[11]=(_Float16)x2.w;
  a[12]=(_Float16)x3.x; a[13]=(_Float16)x3.y; a[14]=(_Float16)x3.z; a[15]=(_Float16)x3.w;
  return a;
}
// --- A fragment, generic gather (functors are branchless clamp+select) ----
template<typename F>
__device__ inline v16h load_a_g(const F& f, int m0, int k0, int lane) {
  const int row = m0 + (lane & 15);
  const int kb  = k0 + ((lane >> 4) << 3);
  v16h a;
#pragma unroll
  for (int e = 0; e < 8; ++e) {
    a[e]     = (_Float16)f(row, kb + e);
    a[e + 8] = (_Float16)f(row, kb + 16 + e);
  }
  return a;
}
// --- stage weights as f16 fragments in per-lane WMMA order ----------------
// B 32x16: lane half h: elems 0..7 -> K=h*8+e, 8..15 -> K=16+h*8+e, col = lane&15.
template<int NT, int KS, typename FB>
__device__ inline void stage_b(_Float16* bst, const FB& fb) {
  for (int i = threadIdx.x; i < NT * KS * 512; i += NTHREADS) {
    const int frag = i >> 9;
    const int lane = (i >> 4) & 31;
    const int e    = i & 15;
    const int nt = frag / KS, ks = frag - nt * KS;
    const int k = ks * 32 + ((lane >> 4) << 3) + (e & 7) + ((e >> 3) << 4);
    const int n = (nt << 4) + (lane & 15);
    bst[i] = (_Float16)fb(k, n);
  }
}

// --- dense GEMM: A row-major f32 [m][32], K=32 ----------------------------
template<int MT, int NT, typename FB, typename FE>
__device__ inline void gemm_d(_Float16* bst, const float* A,
                              const FB& fb, const FE& epi) {
  stage_b<NT, 1>(bst, fb);
  __syncthreads();
  const int lane = threadIdx.x & 31;
  const int wave = threadIdx.x >> 5;
  for (int tile = wave; tile < MT * NT; tile += NW) {
    const int mt = tile / NT, nt = tile - mt * NT;
    const v16h av = load_a_rm(A, mt << 4, lane);
    const v16h bv = *(const v16h*)(bst + (nt << 9) + (lane << 4));
    v8f acc = {0.f,0.f,0.f,0.f,0.f,0.f,0.f,0.f};
    acc = __builtin_amdgcn_wmma_f32_16x16x32_f16(false, av, false, bv,
                                                 (short)0, acc, false, false);
    const int mb = (mt << 4) + ((lane >> 4) << 3);
    const int n  = (nt << 4) + (lane & 15);
#pragma unroll
    for (int r = 0; r < 8; ++r) epi(mb + r, n, acc[r]);
  }
  __syncthreads();
}
// --- generic GEMM: A via gather functor, K = KS*32 ------------------------
template<int MT, int NT, int KS, typename FA, typename FB, typename FE>
__device__ inline void gemm_g(_Float16* bst, const FA& fa,
                              const FB& fb, const FE& epi) {
  stage_b<NT, KS>(bst, fb);
  __syncthreads();
  const int lane = threadIdx.x & 31;
  const int wave = threadIdx.x >> 5;
  for (int tile = wave; tile < MT * NT; tile += NW) {
    const int mt = tile / NT, nt = tile - mt * NT;
    const int m0 = mt << 4;
    v8f acc = {0.f,0.f,0.f,0.f,0.f,0.f,0.f,0.f};
#pragma unroll
    for (int ks = 0; ks < KS; ++ks) {
      const v16h av = load_a_g(fa, m0, ks * 32, lane);
      const v16h bv = *(const v16h*)(bst + ((nt * KS + ks) << 9) + (lane << 4));
      acc = __builtin_amdgcn_wmma_f32_16x16x32_f16(false, av, false, bv,
                                                   (short)0, acc, false, false);
    }
    const int mb = m0 + ((lane >> 4) << 3);
    const int n  = (nt << 4) + (lane & 15);
#pragma unroll
    for (int r = 0; r < 8; ++r) epi(mb + r, n, acc[r]);
  }
  __syncthreads();
}

// --- per-head l2norm + tiny attention (HD=4) ------------------------------
__device__ inline void headnorm(float* buf) {
  for (int g = threadIdx.x; g < PP * 8; g += NTHREADS) {
    const int p = g >> 3, h = g & 7;
    float* v = buf + p * CCH + h * 4;
    const float n = sqrtf(v[0]*v[0] + v[1]*v[1] + v[2]*v[2] + v[3]*v[3]);
    const float inv = 1.0f / fmaxf(n, 1e-12f);
    v[0]*=inv; v[1]*=inv; v[2]*=inv; v[3]*=inv;
  }
}
__device__ inline void attention(const float* Q, const float* K,
                                 const float* V, float* O) {
  for (int task = threadIdx.x; task < TT * 8 * LL; task += NTHREADS) {
    const int l = task % LL;
    const int th = task / LL;
    const int h = th & 7, t = th >> 3;
    const float* qp = Q + (t * LL + l) * CCH + h * 4;
    const float q0=qp[0], q1=qp[1], q2=qp[2], q3=qp[3];
    float s[LL];
    float mx = -1e30f;
#pragma unroll
    for (int m = 0; m < LL; ++m) {
      const float* kp = K + (t * LL + m) * CCH + h * 4;
      const float d = 0.5f * (q0*kp[0] + q1*kp[1] + q2*kp[2] + q3*kp[3]);
      s[m] = d; mx = fmaxf(mx, d);
    }
    float sum = 0.f;
#pragma unroll
    for (int m = 0; m < LL; ++m) { const float e = __expf(s[m] - mx); s[m] = e; sum += e; }
    const float inv = 1.0f / sum;
    float o0=0,o1=0,o2=0,o3=0;
#pragma unroll
    for (int m = 0; m < LL; ++m) {
      const float* vp = V + (t * LL + m) * CCH + h * 4;
      const float w = s[m] * inv;
      o0 += w*vp[0]; o1 += w*vp[1]; o2 += w*vp[2]; o3 += w*vp[3];
    }
    float* op = O + (t * LL + l) * CCH + h * 4;
    op[0]=o0; op[1]=o1; op[2]=o2; op[3]=o3;
  }
}

// ---------------------------------------------------------------------------
__global__ __launch_bounds__(NTHREADS)
void giprb_kernel(Params P) {
  extern __shared__ float smem[];
  float* S0 = smem;
  float* S1 = smem + 1 * SLOT;
  float* S2 = smem + 2 * SLOT;
  float* S3 = smem + 3 * SLOT;
  float* S4 = smem + 4 * SLOT;
  float* S5 = smem + 5 * SLOT;
  float* S6 = smem + 6 * SLOT;                  // shortcut (whole kernel)
  _Float16* bst = (_Float16*)(smem + 7 * SLOT);

  const int fl = P.flag[0];
  const int lo = (fl == 1) ? 18 : (fl == 2) ? 12 : (fl == 3) ? 6 : 0;
  __builtin_prefetch(P.c2w0, 0, 0);
  __builtin_prefetch(P.c2w1, 0, 0);

  // phase 0: async-copy x[b] (c-major) into S1 raw, then transpose to
  // S0 [p][c]; zero pad rows of all slots.
  const float* gx = P.x + (size_t)blockIdx.x * (CCH * PP);
  {
    const unsigned l0 = lds_off(S1);
    for (int i = threadIdx.x; i < (CCH * PP) / 4; i += NTHREADS)
      async_g2l_b128(l0 + (unsigned)i * 16u, (const char*)gx + (size_t)i * 16u);
    wait_async0();
  }
  __syncthreads();
  for (int i = threadIdx.x; i < CCH * PP; i += NTHREADS) {
    const int c = i / PP, p = i - c * PP;
    S0[p * CCH + c] = S1[i];
  }
  for (int i = threadIdx.x; i < 7 * 256; i += NTHREADS) {
    const int s = i >> 8, o = i & 255;
    smem[s * SLOT + PP * CCH + o] = 0.f;
  }
  __syncthreads();

  // phase 1: shortcut = downsample 1x1 (affine) -> S6 (persistent).
  gemm_d<14, 2>(bst, S0,
    [&](int k, int n) { return P.dsw[n * CCH + k]; },
    [&](int m, int n, float a) {
      S6[m * CCH + n] = (a + P.dsb[n]) * P.dsg[n] + P.dsbeta[n];
    });

  // phase 2: ZN = l2norm over channels -> S2 (vectorized row ops).
  for (int p = threadIdx.x; p < PP; p += NTHREADS) {
    const float4* xr = (const float4*)(S0 + p * CCH);
    float4 v[8]; float ss = 0.f;
#pragma unroll
    for (int j = 0; j < 8; ++j) {
      v[j] = xr[j];
      ss += v[j].x*v[j].x + v[j].y*v[j].y + v[j].z*v[j].z + v[j].w*v[j].w;
    }
    const float inv = 1.0f / fmaxf(sqrtf(ss), 1e-12f);
    float4* zr = (float4*)(S2 + p * CCH);
#pragma unroll
    for (int j = 0; j < 8; ++j) {
      float4 o; o.x=v[j].x*inv; o.y=v[j].y*inv; o.z=v[j].z*inv; o.w=v[j].w*inv;
      zr[j] = o;
    }
  }
  __syncthreads();

  // phase 3: lp1 (32->64) -> G=S0, Z2=S1.
  gemm_d<14, 4>(bst, S2,
    [&](int k, int n) { return P.lp1w[k * 64 + n]; },
    [&](int m, int n, float a) {
      const float v = a + P.lp1b[n];
      if (n < 32) S0[m * CCH + n] = v; else S1[m * CCH + (n - 32)] = v;
    });

  // phase 4: conv1 pair on g (NCHW: C=t=9, H=l=24, W=c=32), chunked.
  for (int i = threadIdx.x; i < SLOT; i += NTHREADS) S3[i] = 0.f;   // ACC
  __syncthreads();
  for (int chunk = 0; chunk < 5; ++chunk) {
    const int co0 = chunk * 8;
    const int cw = (36 - co0 < 8) ? (36 - co0) : 8;
    gemm_g<48, 1, 3>(bst,
      [&](int q, int k) {                        // im2col from G=S0
        const int ti = k / 9, r = k - ti * 9;
        const int hh = (q >> 5) + r / 3 - 1;
        const int ww = (q & 31) + r % 3 - 1;
        const bool ok = (k < 81) & ((unsigned)hh < LL) & ((unsigned)ww < CCH);
        const float v = S0[ok ? (ti * LL + hh) * CCH + ww : 0];
        return ok ? v : 0.f;
      },
      [&](int k, int n) {
        const int ti = k / 9, r = k - ti * 9;
        return (k < 81 && n < cw) ? P.c1w0[((co0 + n) * TT + ti) * 9 + r] : 0.f;
      },
      [&](int q, int n, float a) {               // MID=S2 [cj][q]
        if (n < cw) {
          const int co = co0 + n;
          const float v = (a + P.c1b0[co]) * P.c1g0[co] + P.c1beta0[co];
          S2[n * 768 + q] = fmaxf(v, 0.f);
        }
      });
    gemm_g<48, 1, 3>(bst,
      [&](int q, int k) {                        // im2col from MID=S2
        const int cj = k / 9, r = k - cj * 9;
        const int hh = (q >> 5) + r / 3 - 1;
        const int ww = (q & 31) + r % 3 - 1;
        const bool ok = (cj < cw) & ((unsigned)hh < LL) & ((unsigned)ww < CCH);
        const float v = S2[ok ? cj * 768 + (hh << 5) + ww : 0];
        return ok ? v : 0.f;
      },
      [&](int k, int n) {
        const int cj = k / 9, r = k - cj * 9;
        return (n < TT && cj < cw) ? P.c1w1[(n * 36 + (co0 + cj)) * 9 + r] : 0.f;
      },
      [&](int q, int n, float a) { if (n < TT) S3[n * 768 + q] += a; });
  }
  // m-linear: Gm=S2 = relu(affine(ACC)) @ mw + mb.
  gemm_g<14, 2, 1>(bst,
    [&](int m, int k) {
      const int t = m / LL, l = m - (m / LL) * LL;
      const int tc = (t < TT) ? t : TT - 1;      // pad rows clamp params
      const float v = (S3[tc * 768 + (l << 5) + k] + P.c1b1[tc]) * P.c1g1[tc]
                      + P.c1beta1[tc];
      return fmaxf(v, 0.f);
    },
    [&](int k, int n) { return P.mw[k * CCH + n]; },
    [&](int m, int n, float a) { S2[m * CCH + n] = a + P.mb[n]; });
  // live: S1=Z2, S2=Gm, S6=SC

  auto projF = [&](const float* src, const float* W, const float* bias, float* dst) {
    gemm_d<14, 2>(bst, src,
      [&](int k, int n) { return W[k * CCH + n]; },
      [&](int m, int n, float a) { dst[m * CCH + n] = a + bias[n]; });
  };

  // phase 5: pred_intra.
  projF(S1, P.aqw, P.aqb, S3);
  projF(S1, P.akw, P.akb, S4);
  projF(S1, P.avw, P.avb, S5);
  headnorm(S3); headnorm(S4); headnorm(S5);
  __syncthreads();
  attention(S3, S4, S5, S0);                     // O -> S0
  __syncthreads();
  projF(S0, P.amw, P.amb, S3);                   // XA1 -> P1=S3
  for (int i = threadIdx.x; i < PP * CCH; i += NTHREADS) {  // CON -> S0
    const int p = i >> 5, c = i & 31;
    const int t = p / LL, l = p - t * LL;
    const bool pr = (l >= lo) & (l < lo + 6);
    S0[i] = pr ? P.aprompt[(c * TT + t) * LL + l] : S3[i];
  }
  __syncthreads();
  gemm_g<14, 2, 9>(bst,                          // p_conv -> PR1=S4
    [&](int m, int k) {
      const int ci = k / 9, r = k - ci * 9;
      const int t = m / LL, l = m - (m / LL) * LL;
      const int t2 = t + r / 3 - 1, l2 = l + r % 3 - 1;
      const bool ok = (m < PP) & ((unsigned)t2 < TT) & ((unsigned)l2 < LL);
      const float v = S0[ok ? (t2 * LL + l2) * CCH + ci : 0];
      return ok ? v : 0.f;
    },
    [&](int k, int n) {
      const int ci = k / 9, r = k - ci * 9;
      return P.apcw[(n * CCH + ci) * 9 + r];
    },
    [&](int m, int n, float a) {
      const float v = (a + P.apcb[n]) * P.apcg[n] + P.apcbeta[n];
      S4[m * CCH + n] = fmaxf(v, 0.f);
    });
  gemm_g<18, 1, 1>(bst,                          // p_lin (L->6) -> PRL=S5
    [&](int m, int k) {
      const int c = m / TT, t = m - (m / TT) * TT;
      const bool ok = (k < LL);
      const float v = S4[ok ? (t * LL + k) * CCH + c : 0];
      return ok ? v : 0.f;
    },
    [&](int k, int n) { return (k < LL && n < 6) ? P.aplw[k * 6 + n] : 0.f; },
    [&](int m, int n, float a) {
      S5[m * 16 + n] = a + P.aplb[(n < 6) ? n : 5];
    });
  for (int i = threadIdx.x; i < 288 * 6; i += NTHREADS) {   // P1 = tar - pr
    const int m = i / 6, j = i - (i / 6) * 6;
    const int c = m / TT, t = m - (m / TT) * TT;
    S3[(t * LL + lo + j) * CCH + c] -= S5[m * 16 + j];
  }
  __syncthreads();
  // live: S1=Z2, S2=Gm, S3=P1, S6=SC

  // phase 6: pred_inter.
  projF(S1, P.iqw, P.iqb, S0);
  projF(S1, P.ikw, P.ikb, S4);
  projF(S1, P.ivw, P.ivb, S5);
  headnorm(S0); headnorm(S4); headnorm(S5);
  __syncthreads();
  attention(S0, S4, S5, S1);                     // Z2 dead; O -> S1
  __syncthreads();
  projF(S1, P.imw, P.imb, S0);                   // XA2 -> P2=S0
  for (int i = threadIdx.x; i < PP * CCH; i += NTHREADS) {  // CON2 -> S4
    const int p = i >> 5, c = i & 31;
    const int t = p / LL, l = p - t * LL;
    S4[i] = (t < 8) ? S0[i] : P.iprompt[c * LL + l];
  }
  __syncthreads();
  gemm_g<2, 2, 9>(bst,                           // p_conv (9,1) -> PRC=S5
    [&](int m, int k) {
      const int ci = k / 9, dt = k - ci * 9;
      const bool ok = (m < LL);
      const float v = S4[ok ? (dt * LL + m) * CCH + ci : 0];
      return ok ? v : 0.f;
    },
    [&](int k, int n) {
      const int ci = k / 9, dt = k - ci * 9;
      return P.ipcw[(n * CCH + ci) * 9 + dt];
    },
    [&](int m, int n, float a) {
      const float v = (a + P.ipcb[n]) * P.ipcg[n] + P.ipcbeta[n];
      S5[m * CCH + n] = fmaxf(v, 0.f);
    });
  gemm_g<2, 2, 1>(bst,                           // p_lin (24->24) -> PRL2=S4
    [&](int m, int k) {
      const bool ok = (k < LL);
      const float v = S5[ok ? k * CCH + m : 0];
      return ok ? v : 0.f;
    },
    [&](int k, int n) { return (k < LL && n < LL) ? P.iplw[k * LL + n] : 0.f; },
    [&](int m, int n, float a) {
      S4[m * CCH + n] = a + P.iplb[(n < LL) ? n : LL - 1];
    });
  for (int i = threadIdx.x; i < CCH * LL; i += NTHREADS) {  // P2[t=8]
    const int c = i / LL, l = i - (i / LL) * LL;
    S0[(8 * LL + l) * CCH + c] = S0[(2 * LL + l) * CCH + c] - S4[c * CCH + l];
  }
  __syncthreads();
  // live: S0=P2, S2=Gm, S3=P1, S6=SC

  // phase 7: Z3 = lp2( gelu(Gm) * (P1 + P2) ) -> S4.
  gemm_g<14, 2, 1>(bst,
    [&](int m, int k) {
      const float g = S2[m * CCH + k];
      const float ge = 0.5f * g * (1.0f + erff(g * 0.70710678118654752f));
      return ge * (S3[m * CCH + k] + S0[m * CCH + k]);
    },
    [&](int k, int n) { return P.lp2w[k * CCH + n]; },
    [&](int m, int n, float a) { S4[m * CCH + n] = a + P.lp2b[n]; });

  // phase 8: conv2 pair (32->128->32) chunked + shortcut.
  for (int i = threadIdx.x; i < SLOT; i += NTHREADS) S5[i] = 0.f;   // ACC
  __syncthreads();
  for (int c0 = 0; c0 < 128; c0 += 32) {
    gemm_g<14, 2, 9>(bst,                        // conv2[0] -> MID=S1
      [&](int m, int k) {
        const int ci = k / 9, r = k - ci * 9;
        const int t = m / LL, l = m - (m / LL) * LL;
        const int t2 = t + r / 3 - 1, l2 = l + r % 3 - 1;
        const bool ok = (m < PP) & ((unsigned)t2 < TT) & ((unsigned)l2 < LL);
        const float v = S4[ok ? (t2 * LL + l2) * CCH + ci : 0];
        return ok ? v : 0.f;
      },
      [&](int k, int n) {
        const int ci = k / 9, r = k - ci * 9;
        return P.c2w0[((c0 + n) * CCH + ci) * 9 + r];
      },
      [&](int m, int n, float a) {
        const int co = c0 + n;
        const float v = (a + P.c2b0[co]) * P.c2g0[co] + P.c2beta0[co];
        S1[m * CCH + n] = fmaxf(v, 0.f);
      });
    gemm_g<14, 2, 9>(bst,                        // conv2[1] partial -> ACC=S5
      [&](int m, int k) {
        const int cj = k / 9, r = k - cj * 9;
        const int t = m / LL, l = m - (m / LL) * LL;
        const int t2 = t + r / 3 - 1, l2 = l + r % 3 - 1;
        const bool ok = (m < PP) & ((unsigned)t2 < TT) & ((unsigned)l2 < LL);
        const float v = S1[ok ? (t2 * LL + l2) * CCH + cj : 0];
        return ok ? v : 0.f;
      },
      [&](int k, int n) {
        const int cj = k / 9, r = k - cj * 9;
        return P.c2w1[(n * 128 + (c0 + cj)) * 9 + r];
      },
      [&](int m, int n, float a) { S5[m * CCH + n] += a; });
  }
  // Z4 = relu(affine(ACC)) + shortcut(S6) -> S1.
  for (int i = threadIdx.x; i < PP * CCH; i += NTHREADS) {
    const int n = i & 31;
    const float v = (S5[i] + P.c2b1[n]) * P.c2g1[n] + P.c2beta1[n];
    S1[i] = fmaxf(v, 0.f) + S6[i];
  }
  __syncthreads();
  // live: S0=P2, S1=Z4, S3=P1

  // phase 9: fusion attention + residual; result staged c-major in LDS and
  // streamed out with async LDS->global stores.
  projF(S1, P.fqw, P.fqb, S2);                   // Q from Z4
  projF(S3, P.fkw, P.fkb, S4);                   // K from P1
  projF(S0, P.fvw, P.fvb, S5);                   // V from P2
  headnorm(S2); headnorm(S4); headnorm(S5);
  __syncthreads();
  attention(S2, S4, S5, S3);                     // O -> S3
  __syncthreads();
  gemm_d<14, 2>(bst, S3,                         // OUT (c-major) -> S4
    [&](int k, int n) { return P.fmw[k * CCH + n]; },
    [&](int m, int n, float a) {
      if (m < PP) S4[n * PP + m] = a + P.fmb[n] + S1[m * CCH + n];
    });
  float* gout = P.out + (size_t)blockIdx.x * (CCH * PP);
  {
    const unsigned l0 = lds_off(S4);
    for (int i = threadIdx.x; i < (CCH * PP) / 4; i += NTHREADS)
      async_l2g_b128((char*)gout + (size_t)i * 16u, l0 + (unsigned)i * 16u);
    wait_async0();
  }
}

// ---------------------------------------------------------------------------
// Host launch. Input order: setup_inputs() dict-insertion order, recursing
// into params (lin = {w,b}, cna = {w,b,gamma,beta}):
//   0:x | downsample | lp1 lp2 m | intra(q,k,v,m,prompt,p_conv,p_lin) |
//   inter(same) | conv1[0..1] | conv2[0..1] | fusion(q,k,v,m) | 65:atten_flag
// ---------------------------------------------------------------------------
extern "C" void kernel_launch(void* const* d_in, const int* in_sizes, int n_in,
                              void* d_out, int out_size, void* d_ws, size_t ws_size,
                              hipStream_t stream) {
  (void)d_ws; (void)ws_size; (void)n_in; (void)out_size;
#define FP(i) ((const float*)d_in[(i)])
  Params P;
  P.x = FP(0);
  P.dsw = FP(1);  P.dsb = FP(2);  P.dsg = FP(3);  P.dsbeta = FP(4);
  P.lp1w = FP(5); P.lp1b = FP(6);
  P.lp2w = FP(7); P.lp2b = FP(8);
  P.mw = FP(9);   P.mb = FP(10);
  P.aqw = FP(11); P.aqb = FP(12); P.akw = FP(13); P.akb = FP(14);
  P.avw = FP(15); P.avb = FP(16); P.amw = FP(17); P.amb = FP(18);
  P.aprompt = FP(19);
  P.apcw = FP(20); P.apcb = FP(21); P.apcg = FP(22); P.apcbeta = FP(23);
  P.aplw = FP(24); P.aplb = FP(25);
  P.iqw = FP(26); P.iqb = FP(27); P.ikw = FP(28); P.ikb = FP(29);
  P.ivw = FP(30); P.ivb = FP(31); P.imw = FP(32); P.imb = FP(33);
  P.iprompt = FP(34);
  P.ipcw = FP(35); P.ipcb = FP(36); P.ipcg = FP(37); P.ipcbeta = FP(38);
  P.iplw = FP(39); P.iplb = FP(40);
  P.c1w0 = FP(41); P.c1b0 = FP(42); P.c1g0 = FP(43); P.c1beta0 = FP(44);
  P.c1w1 = FP(45); P.c1b1 = FP(46); P.c1g1 = FP(47); P.c1beta1 = FP(48);
  P.c2w0 = FP(49); P.c2b0 = FP(50); P.c2g0 = FP(51); P.c2beta0 = FP(52);
  P.c2w1 = FP(53); P.c2b1 = FP(54); P.c2g1 = FP(55); P.c2beta1 = FP(56);
  P.fqw = FP(57); P.fqb = FP(58); P.fkw = FP(59); P.fkb = FP(60);
  P.fvw = FP(61); P.fvb = FP(62); P.fmw = FP(63); P.fmb = FP(64);
  P.flag = (const int*)d_in[65];
  P.out = (float*)d_out;
#undef FP
  const int B = in_sizes[0] / (CCH * TT * LL);
  giprb_kernel<<<dim3(B), dim3(NTHREADS), SMEM_BYTES, stream>>>(P);
}
